// hydra_channel_attention_6004364279993
// MI455X (gfx1250) — matrix operand order repaired and checked by reading, not compiled, not run on hardware
//
#include <hip/hip_runtime.h>
#include <hip/hip_bf16.h>

typedef __attribute__((ext_vector_type(16))) _Float16 v16h;
typedef __attribute__((ext_vector_type(8)))  _Float16 v8h;
typedef __attribute__((ext_vector_type(8)))  float    v8f;

typedef unsigned int u32x4 __attribute__((ext_vector_type(4)));
typedef int          i32x4 __attribute__((ext_vector_type(4)));
typedef int          i32x8 __attribute__((ext_vector_type(8)));

// ---- CDNA5 TDM availability (guarded; falls back to manual staging) -------
#if defined(__gfx1250__) && __has_builtin(__builtin_amdgcn_tensor_load_to_lds) && \
    __has_builtin(__builtin_amdgcn_s_wait_tensorcnt)
#define USE_TDM 1
#else
#define USE_TDM 0
#endif

#if __has_include(<hip/amd_detail/amd_gfx1250_TDM.h>)
#define TDM_SIX_ARG 1
#else
#define TDM_SIX_ARG 0
#endif

__device__ __forceinline__ v16h combine16(v8h lo, v8h hi) {
    return __builtin_shufflevector(lo, hi, 0,1,2,3,4,5,6,7,8,9,10,11,12,13,14,15);
}

// Block-wide sum over 256 threads (8 wave32s). All threads get the result.
__device__ __forceinline__ float block_sum(float v) {
    __shared__ float sm[8];
    const int lane = threadIdx.x & 31;
    const int wid  = threadIdx.x >> 5;
#pragma unroll
    for (int o = 16; o > 0; o >>= 1) v += __shfl_xor(v, o, 32);
    if (lane == 0) sm[wid] = v;
    __syncthreads();
    float r = (lane < 8) ? sm[lane] : 0.0f;
#pragma unroll
    for (int o = 4; o > 0; o >>= 1) r += __shfl_xor(r, o, 32);
    r = __shfl(r, 0, 32);
    __syncthreads();   // sm safe for reuse by a later block_sum call
    return r;
}

// ---------------------------------------------------------------------------
// Small helper kernels
// ---------------------------------------------------------------------------
__global__ void __launch_bounds__(256) f32_to_f16_kernel(
    const float* __restrict__ in, _Float16* __restrict__ out, int n) {
    int i = blockIdx.x * blockDim.x + threadIdx.x;
    const int stride = gridDim.x * blockDim.x;
    for (; i < n; i += stride) out[i] = (_Float16)in[i];
}

__global__ void __launch_bounds__(256) zero_f32_kernel(float* __restrict__ p, int n) {
    int i = blockIdx.x * blockDim.x + threadIdx.x;
    if (i < n) p[i] = 0.0f;
}

// ---------------------------------------------------------------------------
// LayerNorm(x) over C=768 -> f16. One block (256 thr) per token; 3 elems/thr.
// ---------------------------------------------------------------------------
__global__ void __launch_bounds__(256) ln_in_kernel(
    const float* __restrict__ x, const float* __restrict__ g,
    const float* __restrict__ b, _Float16* __restrict__ h) {
    const long token = blockIdx.x;
    const float* xr = x + token * 768;
    const int t = threadIdx.x;
    float v0 = xr[t], v1 = xr[t + 256], v2 = xr[t + 512];
    const float mu = block_sum(v0 + v1 + v2) * (1.0f / 768.0f);
    const float d0 = v0 - mu, d1 = v1 - mu, d2 = v2 - mu;
    const float var = block_sum(d0*d0 + d1*d1 + d2*d2) * (1.0f / 768.0f);
    const float rstd = rsqrtf(var + 1e-5f);
    _Float16* hr = h + token * 768;
    hr[t      ] = (_Float16)(d0 * rstd * g[t      ] + b[t      ]);
    hr[t + 256] = (_Float16)(d1 * rstd * g[t + 256] + b[t + 256]);
    hr[t + 512] = (_Float16)(d2 * rstd * g[t + 512] + b[t + 512]);
}

// ---------------------------------------------------------------------------
// kv[b,e] += (k / max(||k||,eps)) * v summed over tokens (atomic f32, 8x768).
// ---------------------------------------------------------------------------
__global__ void __launch_bounds__(256) kv_kernel(
    const _Float16* __restrict__ qkv, float* __restrict__ kv, int tokens_per_batch) {
    const long token = blockIdx.x;
    const int bidx = (int)(token / tokens_per_batch);
    const _Float16* row = qkv + token * 2304;
    const int t = threadIdx.x;
    const float k0 = (float)row[768 + t      ];
    const float k1 = (float)row[768 + t + 256];
    const float k2 = (float)row[768 + t + 512];
    const float ss = block_sum(k0*k0 + k1*k1 + k2*k2);
    const float rn = 1.0f / fmaxf(sqrtf(ss), 1e-12f);
    float* kvb = kv + bidx * 768;
    atomicAdd(&kvb[t      ], k0 * rn * (float)row[1536 + t      ]);
    atomicAdd(&kvb[t + 256], k1 * rn * (float)row[1536 + t + 256]);
    atomicAdd(&kvb[t + 512], k2 * rn * (float)row[1536 + t + 512]);
}

// ---------------------------------------------------------------------------
// attn = (q/||q||) * kv[b]; LayerNorm over E -> f16 activation for proj GEMM.
// ---------------------------------------------------------------------------
__global__ void __launch_bounds__(256) attn_ln_kernel(
    const _Float16* __restrict__ qkv, const float* __restrict__ kv,
    const float* __restrict__ g, const float* __restrict__ b,
    _Float16* __restrict__ a, int tokens_per_batch) {
    const long token = blockIdx.x;
    const int bidx = (int)(token / tokens_per_batch);
    const _Float16* row = qkv + token * 2304;
    const int t = threadIdx.x;
    const float q0 = (float)row[t], q1 = (float)row[t + 256], q2 = (float)row[t + 512];
    const float ss = block_sum(q0*q0 + q1*q1 + q2*q2);
    const float rn = 1.0f / fmaxf(sqrtf(ss), 1e-12f);
    const float* kvb = kv + bidx * 768;
    const float a0 = q0 * rn * kvb[t];
    const float a1 = q1 * rn * kvb[t + 256];
    const float a2 = q2 * rn * kvb[t + 512];
    const float mu = block_sum(a0 + a1 + a2) * (1.0f / 768.0f);
    const float d0 = a0 - mu, d1 = a1 - mu, d2 = a2 - mu;
    const float var = block_sum(d0*d0 + d1*d1 + d2*d2) * (1.0f / 768.0f);
    const float rstd = rsqrtf(var + 1e-5f);
    _Float16* ar = a + token * 768;
    ar[t      ] = (_Float16)(d0 * rstd * g[t      ] + b[t      ]);
    ar[t + 256] = (_Float16)(d1 * rstd * g[t + 256] + b[t + 256]);
    ar[t + 512] = (_Float16)(d2 * rstd * g[t + 512] + b[t + 512]);
}

// ---------------------------------------------------------------------------
// f16 WMMA GEMM: D[M,N] = A[M,K] * B[K,N], f32 accumulate.
// Workgroup tile 128x128, K-step 32, double-buffered LDS, one barrier/step.
// 8 waves (4x2); each wave: 2x4 tiles of v_wmma_f32_16x16x32_f16 per step.
// A tile staged by the Tensor Data Mover (when available); B staged through
// registers with a transposing packed-b32 LDS scatter.
// ---------------------------------------------------------------------------
#define GT_M 128
#define GT_N 128
#define GT_K 32
#define LDH  40   // padded LDS row stride in halves (80B, keeps 16B alignment)

#if USE_TDM
// Issue one TDM 2D tile load: rows x 32-halves tile of a row-major [rows_total x
// row_len] f16 tensor -> padded LDS rows (64B data + 16B pad == LDH halves).
// D# bit layout per CDNA5 ISA 8.3-8.6.
__device__ __forceinline__ void tdm_load_a_tile(const _Float16* gsrc, void* lds_dst,
                                                int row_len, int rows_total) {
    const unsigned long long ga = (unsigned long long)(const void*)gsrc;
    const unsigned int lds_off = (unsigned int)(unsigned long long)(size_t)lds_dst;
    u32x4 g0;
    g0[0] = 1u;                                          // count=1, user descriptor
    g0[1] = lds_off;                                     // lds_addr
    g0[2] = (unsigned int)(ga & 0xFFFFFFFFu);            // global_addr[31:0]
    g0[3] = (unsigned int)((ga >> 32) & 0x1FFFFFFu)      // global_addr[56:32]
          | (2u << 30);                                  // type=2 ("image")
    i32x8 g1;
    g1[0] = (1 << 16)                                    // data_size = 2 bytes
          | (1 << 20)                                    // pad_enable
          | (3 << 22)                                    // pad_interval: 16 DWORDs (64B)
          | (3 << 25);                                   // pad_amount: 4 DWORDs (16B)
    g1[1] = (int)((row_len & 0xFFFF) << 16);             // tensor_dim0[15:0]
    g1[2] = (int)(((unsigned)row_len >> 16) & 0xFFFF)    // tensor_dim0[31:16]
          | (int)((rows_total & 0xFFFF) << 16);          // tensor_dim1[15:0]
    g1[3] = (int)(((unsigned)rows_total >> 16) & 0xFFFF) // tensor_dim1[31:16]
          | (GT_K << 16);                                // tile_dim0 = 32
    g1[4] = GT_M;                                        // tile_dim1 = 128 (tile_dim2=0)
    g1[5] = row_len;                                     // tensor_dim0_stride[31:0]
    g1[6] = 0;                                           // stride hi / dim1_stride lo
    g1[7] = 0;
    const i32x4 z4 = {0, 0, 0, 0};
#if TDM_SIX_ARG
    const i32x8 z8 = {0, 0, 0, 0, 0, 0, 0, 0};
    __builtin_amdgcn_tensor_load_to_lds(g0, g1, z4, z4, z8, 0);
#else
    __builtin_amdgcn_tensor_load_to_lds(g0, g1, z4, z4, 0);
#endif
}
#endif  // USE_TDM

__device__ __forceinline__ v16h load_afrag(const _Float16* as, int row, int kbA) {
    const _Float16* p = &as[row * LDH];
    v8h lo = *(const v8h*)&p[kbA];
    v8h hi = *(const v8h*)&p[kbA + 16];
    return combine16(lo, hi);
}
__device__ __forceinline__ v16h load_bfrag(const _Float16* bs, int col, int kbB) {
    const _Float16* p = &bs[col * LDH + kbB];
    v8h lo = *(const v8h*)&p[0];
    v8h hi = *(const v8h*)&p[16];
    return combine16(lo, hi);
}
// Transposing scatter of two adjacent K-rows: pack (k,k+1) into one b32 store.
__device__ __forceinline__ void store_b_pair(_Float16* bs, int bn, int bk2, v8h lo, v8h hi) {
#pragma unroll
    for (int j = 0; j < 8; ++j) {
        union { _Float16 h[2]; unsigned int u; } p;
        p.h[0] = lo[j];
        p.h[1] = hi[j];
        *(unsigned int*)&bs[(bn + j) * LDH + bk2] = p.u;
    }
}

template <typename OutT>
__global__ void __launch_bounds__(256) gemm_f16_wmma(
    const _Float16* __restrict__ A, const _Float16* __restrict__ B,
    OutT* __restrict__ D, int M, int N, int K) {
    __shared__ _Float16 As[2][GT_M * LDH];   // [row][k] padded
    __shared__ _Float16 Bs[2][GT_N * LDH];   // [col][k] (transposed staging)

    const int tid  = threadIdx.x;
    const int lane = tid & 31;
    const int wid  = tid >> 5;
    const int wm   = wid >> 1;                // 0..3 : wave M position
    const int wn   = wid & 1;                 // 0..1 : wave N position
    const int m0   = blockIdx.y * GT_M;
    const int n0   = blockIdx.x * GT_N;

    const v8f vzero = {0.f, 0.f, 0.f, 0.f, 0.f, 0.f, 0.f, 0.f};
    v8f acc[2][4];
#pragma unroll
    for (int i = 0; i < 2; ++i)
#pragma unroll
        for (int j = 0; j < 4; ++j) acc[i][j] = vzero;

    // staging assignments
    const int bk2  = (tid >> 4) * 2;          // adjacent K-row pair 0,2,..,30
    const int bn   = (tid & 15) * 8;          // 0..120
    const int arow = tid >> 2;                // 0..63 (rows arow, arow+64)
    const int acol = (tid & 3) * 8;           // 0,8,16,24
    (void)arow; (void)acol;

    // WMMA fragment geometry (ISA 7.12.2)
    const int mlane = lane & 15;
    const int kbA   = (lane < 16) ? 0 : 8;
    const int kbB   = (lane < 16) ? 0 : 8;
    const int rbase = 8 * (lane >> 4);

    const int ktiles = K / GT_K;

    // ---- prologue: stage tile 0 into buffer 0 ----
    {
        v8h pb0 = *(const v8h*)&B[bk2 * N + n0 + bn];
        v8h pb1 = *(const v8h*)&B[(bk2 + 1) * N + n0 + bn];
#if USE_TDM
        if (wid == 0) tdm_load_a_tile(&A[(size_t)m0 * K], &As[0][0], K, M);
#else
        v8h pa0 = *(const v8h*)&A[(m0 + arow) * K + acol];
        v8h pa1 = *(const v8h*)&A[(m0 + arow + 64) * K + acol];
        *(v8h*)&As[0][arow * LDH + acol]        = pa0;
        *(v8h*)&As[0][(arow + 64) * LDH + acol] = pa1;
#endif
        store_b_pair(&Bs[0][0], bn, bk2, pb0, pb1);
#if USE_TDM
        if (wid == 0) __builtin_amdgcn_s_wait_tensorcnt(0);
#endif
        __syncthreads();
    }

    for (int kt = 0; kt < ktiles; ++kt) {
        const int cur = kt & 1;
        const int nxt = cur ^ 1;
        const bool more = (kt + 1) < ktiles;
        const int k1 = (kt + 1) * GT_K;

        // issue next tile's global traffic before the math (latency hiding)
        v8h nb0{}, nb1{};
#if !USE_TDM
        v8h na0{}, na1{};
#endif
        if (more) {
            nb0 = *(const v8h*)&B[(k1 + bk2) * N + n0 + bn];
            nb1 = *(const v8h*)&B[(k1 + bk2 + 1) * N + n0 + bn];
#if USE_TDM
            if (wid == 0) tdm_load_a_tile(&A[(size_t)m0 * K + k1], &As[nxt][0], K, M);
#else
            na0 = *(const v8h*)&A[(m0 + arow) * K + k1 + acol];
            na1 = *(const v8h*)&A[(m0 + arow + 64) * K + k1 + acol];
            __builtin_prefetch(&A[(m0 + arow) * K + k1 + GT_K + acol], 0, 1);
#endif
        }

        // ---- 8x v_wmma_f32_16x16x32_f16 on the current buffer ----
        v16h af[2], bf[4];
#pragma unroll
        for (int i = 0; i < 2; ++i)
            af[i] = load_afrag(&As[cur][0], wm * 32 + i * 16 + mlane, kbA);
#pragma unroll
        for (int j = 0; j < 4; ++j)
            bf[j] = load_bfrag(&Bs[cur][0], wn * 64 + j * 16 + mlane, kbB);
#pragma unroll
        for (int i = 0; i < 2; ++i)
#pragma unroll
            for (int j = 0; j < 4; ++j)
                acc[i][j] = __builtin_amdgcn_wmma_f32_16x16x32_f16(
                    false, af[i], false, bf[j], (short)0, acc[i][j], false, false);

        // ---- stage next tile into the other buffer ----
        if (more) {
            store_b_pair(&Bs[nxt][0], bn, bk2, nb0, nb1);
#if !USE_TDM
            *(v8h*)&As[nxt][arow * LDH + acol]        = na0;
            *(v8h*)&As[nxt][(arow + 64) * LDH + acol] = na1;
#endif
        }
#if USE_TDM
        if (wid == 0) __builtin_amdgcn_s_wait_tensorcnt(0);
#endif
        __syncthreads();
    }

    // ---- epilogue: D layout per ISA 7.12.2 (VGPR r -> row r + 8*(lane/16)) ----
#pragma unroll
    for (int i = 0; i < 2; ++i) {
#pragma unroll
        for (int j = 0; j < 4; ++j) {
            const int col  = n0 + wn * 64 + j * 16 + mlane;
            const int row0 = m0 + wm * 32 + i * 16 + rbase;
#pragma unroll
            for (int r = 0; r < 8; ++r)
                D[(long)(row0 + r) * N + col] = (OutT)acc[i][j][r];
        }
    }
}

// ---------------------------------------------------------------------------
// Launch
// ---------------------------------------------------------------------------
extern "C" void kernel_launch(void* const* d_in, const int* in_sizes, int n_in,
                              void* d_out, int out_size, void* d_ws, size_t ws_size,
                              hipStream_t stream) {
    (void)in_sizes; (void)n_in; (void)out_size; (void)ws_size;
    const float* x        = (const float*)d_in[0];
    const float* w_qkv    = (const float*)d_in[1];
    const float* w_proj   = (const float*)d_in[2];
    const float* ln_in_g  = (const float*)d_in[3];
    const float* ln_in_b  = (const float*)d_in[4];
    const float* ln_out_g = (const float*)d_in[5];
    const float* ln_out_b = (const float*)d_in[6];

    const int B = 8, N = 4096, C = 768, E = 768;
    const int T = B * N;  // 32768 tokens

    char* ws = (char*)d_ws;
    size_t off = 0;
    auto salloc = [&](size_t bytes) -> void* {
        void* p = ws + off;
        off = (off + bytes + 255) & ~(size_t)255;
        return p;
    };
    _Float16* h16     = (_Float16*)salloc((size_t)T * C * 2);       // LN(x); reused for attn act
    _Float16* qkv16   = (_Float16*)salloc((size_t)T * 3 * E * 2);
    _Float16* wqkv16  = (_Float16*)salloc((size_t)C * 3 * E * 2);
    _Float16* wproj16 = (_Float16*)salloc((size_t)E * C * 2);
    float*    kv      = (float*)   salloc((size_t)B * E * 4);

    f32_to_f16_kernel<<<1024, 256, 0, stream>>>(w_qkv, wqkv16, C * 3 * E);
    f32_to_f16_kernel<<<512,  256, 0, stream>>>(w_proj, wproj16, E * C);
    zero_f32_kernel<<<(B * E + 255) / 256, 256, 0, stream>>>(kv, B * E);
    ln_in_kernel<<<T, 256, 0, stream>>>(x, ln_in_g, ln_in_b, h16);

    dim3 g_qkv(3 * E / GT_N, T / GT_M);   // (18, 256)
    gemm_f16_wmma<_Float16><<<g_qkv, 256, 0, stream>>>(h16, wqkv16, qkv16, T, 3 * E, C);

    kv_kernel<<<T, 256, 0, stream>>>(qkv16, kv, N);
    attn_ln_kernel<<<T, 256, 0, stream>>>(qkv16, kv, ln_out_g, ln_out_b, h16, N);

    dim3 g_proj(C / GT_N, T / GT_M);      // (6, 256)
    gemm_f16_wmma<float><<<g_proj, 256, 0, stream>>>(h16, wproj16, (float*)d_out, T, C, E);
}
